// MLPModularGaussianModule_79869211837085
// MI455X (gfx1250) — compile-verified, matrix operand order checked by reading, not run on hardware
//
#include <hip/hip_runtime.h>

// Problem constants (match reference)
constexpr int NVARS = 128;
constexpr int NMOD  = 128;
constexpr int NHID  = 256;
constexpr int NBATCH = 4096;

constexpr int CHUNK = 512;  // batch per workgroup
constexpr int NB    = 64;   // inner batch tile
constexpr int XS    = 136;  // xm LDS row stride (elems), 272B = 17*16
constexpr int HS    = 264;  // h1/h2 LDS row stride (elems), 528B = 33*16

typedef __attribute__((ext_vector_type(16))) __bf16 v16bf;
typedef __attribute__((ext_vector_type(8)))  float  v8f;

// LDS layout (dynamic shared, byte offsets)
constexpr unsigned OFF_W0  = 0;                          // [256][128] bf16 = 65536
constexpr unsigned OFF_W1  = 65536;                      // [256][256] bf16 = 131072
constexpr unsigned OFF_XM  = 196608;                     // [64][136] bf16  = 17408
constexpr unsigned OFF_H1  = 214016;                     // [64][264] bf16  = 33792
constexpr unsigned OFF_H2  = 247808;                     // [64][264] bf16  = 33792
constexpr unsigned OFF_B0  = 281600;                     // 256 f32
constexpr unsigned OFF_B1  = 282624;                     // 256 f32
constexpr unsigned OFF_W2  = 283648;                     // 256 f32
constexpr unsigned OFF_PT  = 284672;                     // [4][64] f32
constexpr unsigned SMEM_BYTES = 285696;                  // ~279 KB < 320 KB WGP LDS

// Load a 16x32 bf16 fragment from LDS (row-major, 16B-aligned rows).
// Per CDNA5 ISA 16-bit A layout: lane L<16 holds row L, K = 0..7 & 16..23;
// lane L>=16 holds row L-16, K = 8..15 & 24..31. B fragments use the same
// math when the 32x16 B tile is stored transposed ([N][K] row-major).
__device__ __forceinline__ v16bf load_frag(const __bf16* base, int stride) {
  const int lane = threadIdx.x & 31;
  const __bf16* p = base + (lane & 15) * stride + ((lane >> 4) << 3);
  v16bf r;
  ((uint4*)&r)[0] = *(const uint4*)p;          // K = kb .. kb+7
  ((uint4*)&r)[1] = *(const uint4*)(p + 16);   // K = kb+16 .. kb+23
  return r;
}

// One fused GEMM stage: Out[col][row] = lrelu(A[rows][K] @ B^T[cols][K] + bias[row])
// Block = 8 waves; wave w computes rows [w*32, w*32+32) x cols [0,64).
__device__ __forceinline__ void gemm_tile(const __bf16* A, int lda,
                                          const __bf16* B, int ldb, int K,
                                          const float* bias,
                                          __bf16* Out, int ldo) {
  const int tid = threadIdx.x;
  const int w   = tid >> 5;
  const int lane = tid & 31;

  v8f acc[2][4];
#pragma unroll
  for (int r = 0; r < 2; ++r)
#pragma unroll
    for (int c = 0; c < 4; ++c) acc[r][c] = 0.0f;  // splat

  for (int k = 0; k < K; k += 32) {
    v16bf a0 = load_frag(A + (w * 32 + 0)  * lda + k, lda);
    v16bf a1 = load_frag(A + (w * 32 + 16) * lda + k, lda);
    v16bf bf[4];
#pragma unroll
    for (int c = 0; c < 4; ++c) bf[c] = load_frag(B + (c * 16) * ldb + k, ldb);
#pragma unroll
    for (int c = 0; c < 4; ++c) {
      acc[0][c] = __builtin_amdgcn_wmma_f32_16x16x32_bf16(
          false, a0, false, bf[c], (short)0, acc[0][c], false, false);
      acc[1][c] = __builtin_amdgcn_wmma_f32_16x16x32_bf16(
          false, a1, false, bf[c], (short)0, acc[1][c], false, false);
    }
  }

  // Epilogue: C/D layout — VGPR v: lanes 0-15 -> M=v, N=lane; lanes 16-31 -> M=v+8.
  // Lane holds 8 consecutive rows -> bias + lrelu + pack 8 bf16 -> one 16B LDS store.
  const int n  = lane & 15;
  const int mo = (lane >> 4) << 3;
#pragma unroll
  for (int r = 0; r < 2; ++r)
#pragma unroll
    for (int c = 0; c < 4; ++c) {
      const int ib  = w * 32 + r * 16 + mo;
      const int col = c * 16 + n;
      __bf16 tmp[8];
#pragma unroll
      for (int e = 0; e < 8; ++e) {
        float f = acc[r][c][e] + bias[ib + e];
        f = f > 0.0f ? f : 0.01f * f;   // leaky_relu(0.01)
        tmp[e] = (__bf16)f;
      }
      *(uint4*)(Out + col * ldo + ib) = *(uint4*)tmp;
    }
}

// Phase A: per (module t, batch chunk): fused V->HID->HID->1 MLP -> h_mod[b][t]
__global__ __launch_bounds__(256)
void n2m_fused_kernel(const float* __restrict__ x,
                      const float* __restrict__ mask_n2m,
                      const float* __restrict__ W0, const float* __restrict__ b0,
                      const float* __restrict__ W1, const float* __restrict__ b1,
                      const float* __restrict__ W2, const float* __restrict__ b2,
                      float* __restrict__ h_mod) {
  extern __shared__ char smem[];
  __bf16* W0s = (__bf16*)(smem + OFF_W0);
  __bf16* W1s = (__bf16*)(smem + OFF_W1);
  __bf16* xms = (__bf16*)(smem + OFF_XM);
  __bf16* h1s = (__bf16*)(smem + OFF_H1);
  __bf16* h2s = (__bf16*)(smem + OFF_H2);
  float*  b0s = (float*)(smem + OFF_B0);
  float*  b1s = (float*)(smem + OFF_B1);
  float*  w2s = (float*)(smem + OFF_W2);
  float*  part = (float*)(smem + OFF_PT);

  const int t     = blockIdx.x;          // module
  const int bbase = blockIdx.y * CHUNK;  // batch chunk
  const int tid   = threadIdx.x;

  // Stage weights once per WG: fp32 -> bf16 into LDS (coalesced, L2-resident)
  for (int i = tid; i < NHID * NVARS; i += 256)
    W0s[i] = (__bf16)W0[(size_t)t * NHID * NVARS + i];
  for (int i = tid; i < NHID * NHID; i += 256)
    W1s[i] = (__bf16)W1[(size_t)t * NHID * NHID + i];
  for (int i = tid; i < NHID; i += 256) {
    b0s[i] = b0[t * NHID + i];
    b1s[i] = b1[t * NHID + i];
    w2s[i] = W2[t * NHID + i];
  }
  const float bias2 = b2[t];
  __syncthreads();

  for (int bt = 0; bt < CHUNK / NB; ++bt) {
    const int bti = bbase + bt * NB;

    // Build masked input tile xm[b][j] = x[b,j] * mask_n2m[b,j,t] (bf16)
    for (int e = tid; e < NB * NVARS; e += 256) {
      const int bl = e >> 7;             // NVARS == 128
      const int j  = e & (NVARS - 1);
      const size_t row = (size_t)(bti + bl) * NVARS + j;
      xms[bl * XS + j] = (__bf16)(x[row] * mask_n2m[row * NMOD + t]);
    }

    // Prefetch next tile's gather lines (global_prefetch_b8) so the
    // stride-512B mask reads are L0/L2-resident by the time the two GEMMs
    // below finish. Fire-and-forget: no counters, no data return.
    if (bt + 1 < CHUNK / NB) {
      const int btn = bti + NB;
      for (int e = tid; e < NB * NVARS; e += 256) {
        const int bl = e >> 7;
        const int j  = e & (NVARS - 1);
        const size_t row = (size_t)(btn + bl) * NVARS + j;
        __builtin_prefetch(&mask_n2m[row * NMOD + t], 0, 1);
      }
      // x tile for next iteration: 32 KB = 256 lines; one line per thread
      __builtin_prefetch(&x[(size_t)btn * NVARS + tid * 32], 0, 1);
    }
    __syncthreads();

    // layer 0: h1 = lrelu(W0 @ xm + b0)   [256 x 64], K = 128
    gemm_tile(W0s, NVARS, xms, XS, NVARS, b0s, h1s, HS);
    __syncthreads();

    // layer 1: h2 = lrelu(W1 @ h1 + b1)   [256 x 64], K = 256
    gemm_tile(W1s, NHID, h1s, HS, NHID, b1s, h2s, HS);
    __syncthreads();

    // layer 2: scalar head — 4 threads per batch element, partial dots
    {
      const int bl = tid & 63;
      const int seg = tid >> 6;
      const __bf16* hrow = h2s + bl * HS + seg * 64;
      float s = 0.0f;
#pragma unroll 8
      for (int h = 0; h < 64; ++h) s += w2s[seg * 64 + h] * (float)hrow[h];
      part[seg * 64 + bl] = s;
    }
    __syncthreads();
    if (tid < NB) {
      const float s = part[tid] + part[64 + tid] + part[128 + tid] + part[192 + tid] + bias2;
      h_mod[(size_t)(bti + tid) * NMOD + t] = s;
    }
    __syncthreads();
  }
}

// Phase B: out[b,v] = bm[v] + sum_t Wm[v,t] * mask_m2n[b,v,t] * h_mod[b,t]
// One wave per (b,v); lanes stride contiguous t -> fully coalesced streaming.
__global__ __launch_bounds__(256)
void m2n_kernel(const float* __restrict__ mask_m2n,
                const float* __restrict__ Wm, const float* __restrict__ bm,
                const float* __restrict__ h_mod, float* __restrict__ out) {
  const int gtid = blockIdx.x * 256 + threadIdx.x;
  const int gw   = gtid >> 5;
  const int lane = gtid & 31;
  const int b = gw >> 7;            // NVARS == 128
  const int v = gw & (NVARS - 1);

  const float* mrow = mask_m2n + ((size_t)b * NVARS + v) * NMOD;
  const float* hrow = h_mod + (size_t)b * NMOD;
  const float* wrow = Wm + v * NMOD;

  float s = 0.0f;
#pragma unroll
  for (int i = 0; i < NMOD / 32; ++i) {
    const int tt = lane + i * 32;
    s += wrow[tt] * mrow[tt] * hrow[tt];
  }
#pragma unroll
  for (int off = 16; off > 0; off >>= 1) s += __shfl_xor(s, off, 32);
  if (lane == 0) out[(size_t)b * NVARS + v] = s + bm[v];
}

extern "C" void kernel_launch(void* const* d_in, const int* in_sizes, int n_in,
                              void* d_out, int out_size, void* d_ws, size_t ws_size,
                              hipStream_t stream) {
  const float* x        = (const float*)d_in[0];
  const float* mask_n2m = (const float*)d_in[1];
  const float* mask_m2n = (const float*)d_in[2];
  const float* W0 = (const float*)d_in[3];
  const float* b0 = (const float*)d_in[4];
  const float* W1 = (const float*)d_in[5];
  const float* b1 = (const float*)d_in[6];
  const float* W2 = (const float*)d_in[7];
  const float* b2 = (const float*)d_in[8];
  const float* Wm = (const float*)d_in[9];
  const float* bm = (const float*)d_in[10];

  float* out   = (float*)d_out;
  float* h_mod = (float*)d_ws;  // NBATCH * NMOD f32 = 2 MB scratch

  dim3 gridA(NMOD, NBATCH / CHUNK);   // 128 x 8 workgroups
  n2m_fused_kernel<<<gridA, 256, SMEM_BYTES, stream>>>(
      x, mask_n2m, W0, b0, W1, b1, W2, b2, h_mod);

  const int total_waves = NBATCH * NVARS;       // one wave per output element
  const int blocksB = total_waves / 8;          // 8 waves per 256-thread block
  m2n_kernel<<<blocksB, 256, 0, stream>>>(mask_m2n, Wm, bm, h_mod, out);
}
